// MultiHeadSelfAttention_47270410060088
// MI455X (gfx1250) — compile-verified, matrix operand order checked by reading, not run on hardware
//
#include <hip/hip_runtime.h>

// ---------------------------------------------------------------------------
// Problem constants
// ---------------------------------------------------------------------------
constexpr int B  = 4;
constexpr int S  = 2048;
constexpr int F  = 512;
constexpr int H  = 8;
constexpr int BS = B * S;        // 8192
constexpr int HF = H * F;        // 4096

// LDS strides (padded to avoid bank conflicts)
constexpr int SSTR = 2068;       // floats per score row  (2068 % 64 == 20 -> distinct banks over 16 rows)
constexpr int PSTR = 2056;       // ushorts per P row     (1028 words/row, %64 == 4 -> conflict-free b128)

constexpr int SC_BYTES = 16 * SSTR * 4;            // 132352
constexpr int P_OFF    = SC_BYTES;                 // 132352
constexpr int P_BYTES  = 16 * PSTR * 2;            // 65792
constexpr int RED_OFF  = P_OFF + P_BYTES;          // 198144  (256 floats)
constexpr int RM_OFF   = RED_OFF + 256 * 4;        // 199168
constexpr int RS_OFF   = RM_OFF + 16 * 4;          // 199232
constexpr int SMEM_TOT = RS_OFF + 16 * 4;          // 199296 bytes (< 320 KB LDS)

// Workspace layout (bytes)
constexpr size_t XB_OFF  = 0;                                   // x bf16        [BS][F]
constexpr size_t WQT_OFF = XB_OFF  + (size_t)BS * F * 2;        // WqT bf16      [H][F][F]
constexpr size_t WKT_OFF = WQT_OFF + (size_t)H  * F * F * 2;
constexpr size_t WVT_OFF = WKT_OFF + (size_t)H  * F * F * 2;
constexpr size_t WJT_OFF = WVT_OFF + (size_t)H  * F * F * 2;    // WjT bf16      [F][HF]
constexpr size_t QB_OFF  = WJT_OFF + (size_t)F  * HF * 2;       // Q bf16        [B*H][S][F]
constexpr size_t KB_OFF  = QB_OFF  + (size_t)B * H * S * F * 2; // K bf16        [B*H][S][F]
constexpr size_t VT_OFF  = KB_OFF  + (size_t)B * H * S * F * 2; // V^T bf16      [B*H][F][S]
constexpr size_t ZC_OFF  = VT_OFF  + (size_t)B * H * S * F * 2; // zcat bf16     [BS][HF]
constexpr size_t JO_OFF  = ZC_OFF  + (size_t)BS * HF * 2;       // joiner f32    [BS][F]

// ---------------------------------------------------------------------------
// WMMA plumbing (CDNA5 / gfx1250, wave32)
// ---------------------------------------------------------------------------
typedef __bf16 bf16_t;
typedef bf16_t v16bf __attribute__((ext_vector_type(16)));
typedef float  v8f   __attribute__((ext_vector_type(8)));
typedef unsigned int v4u __attribute__((ext_vector_type(4)));

union Frag {
    v16bf v;
    v4u   q[2];
};

__device__ __forceinline__ v16bf load_frag(const unsigned short* p) {
    // A/B^T layout: lane holds K-chunks [base .. base+7] and [base+16 .. base+23]
    Frag f;
    f.q[0] = *reinterpret_cast<const v4u*>(p);
    f.q[1] = *reinterpret_cast<const v4u*>(p + 16);
    return f.v;
}

__device__ __forceinline__ v8f wmma_bf16(v16bf a, v16bf b, v8f c) {
    return __builtin_amdgcn_wmma_f32_16x16x32_bf16(false, a, false, b, (short)0, c, false, false);
}

__device__ __forceinline__ v8f vzero() {
    v8f z;
    #pragma unroll
    for (int i = 0; i < 8; ++i) z[i] = 0.0f;
    return z;
}

__device__ __forceinline__ unsigned short f2bf(float x) {
    unsigned int u = __float_as_uint(x);
    unsigned int r = u + 0x7FFFu + ((u >> 16) & 1u);   // round-to-nearest-even
    return (unsigned short)(r >> 16);
}

// ---------------------------------------------------------------------------
// Stage 0: conversions
// ---------------------------------------------------------------------------
__global__ void cvt_x_kernel(const float* __restrict__ in, unsigned short* __restrict__ out, int n) {
    for (int i = blockIdx.x * blockDim.x + threadIdx.x; i < n; i += gridDim.x * blockDim.x)
        out[i] = f2bf(in[i]);
}

// out[b][c][r] = bf16(in[b][r][c])  -- batched transpose + convert
__global__ void transpose_cvt_kernel(const float* __restrict__ in, unsigned short* __restrict__ out,
                                     int R, int C, int total) {
    for (int idx = blockIdx.x * blockDim.x + threadIdx.x; idx < total; idx += gridDim.x * blockDim.x) {
        int r = idx % R;
        int c = (idx / R) % C;
        int b = idx / (R * C);
        out[idx] = f2bf(in[(size_t)b * R * C + (size_t)r * C + c]);
    }
}

// ---------------------------------------------------------------------------
// Stage 1: QKV projections.  grid = (BS/64, F/64, 3*H), 128 threads (4 waves)
// ---------------------------------------------------------------------------
__global__ __launch_bounds__(128) void qkv_kernel(
        const unsigned short* __restrict__ xb,
        const unsigned short* __restrict__ wqt,
        const unsigned short* __restrict__ wkt,
        const unsigned short* __restrict__ wvt,
        unsigned short* __restrict__ Qb,
        unsigned short* __restrict__ Kb,
        unsigned short* __restrict__ Vt) {
    const int lane = threadIdx.x & 31;
    const int wave = threadIdx.x >> 5;
    const int l16  = lane & 15;
    const int half = lane >> 4;

    const int proj = blockIdx.z / H;        // 0=Q 1=K 2=V
    const int h    = blockIdx.z % H;
    const int m0   = blockIdx.x * 64 + wave * 16;   // flat row in [0,BS)
    const int n0   = blockIdx.y * 64;

    const unsigned short* wt =
        (proj == 0 ? wqt : (proj == 1 ? wkt : wvt)) + (size_t)h * F * F;
    const unsigned short* arow = xb + (size_t)(m0 + l16) * F + half * 8;

    v8f acc[4] = {};
    #pragma unroll 4
    for (int k = 0; k < F / 32; ++k) {
        v16bf a = load_frag(arow + k * 32);
        #pragma unroll
        for (int nt = 0; nt < 4; ++nt) {
            const unsigned short* brow =
                wt + (size_t)(n0 + nt * 16 + l16) * F + k * 32 + half * 8;
            acc[nt] = wmma_bf16(a, load_frag(brow), acc[nt]);
        }
    }

    const int b      = m0 / S;
    const int s_base = m0 % S;
    const int bh     = b * H + h;

    if (proj < 2) {
        unsigned short* out = (proj == 0 ? Qb : Kb) + (size_t)bh * S * F;
        #pragma unroll
        for (int nt = 0; nt < 4; ++nt) {
            int e = n0 + nt * 16 + l16;
            #pragma unroll
            for (int r = 0; r < 8; ++r) {
                int s = s_base + r + 8 * half;
                out[(size_t)s * F + e] = f2bf(acc[nt][r]);
            }
        }
    } else {
        unsigned short* out = Vt + (size_t)bh * F * S;   // V^T: [F][S]
        #pragma unroll
        for (int nt = 0; nt < 4; ++nt) {
            int e = n0 + nt * 16 + l16;
            #pragma unroll
            for (int r = 0; r < 8; ++r) {
                int s = s_base + r + 8 * half;
                out[(size_t)e * S + s] = f2bf(acc[nt][r]);
            }
        }
    }
}

// ---------------------------------------------------------------------------
// Stage 2: attention.  grid = (S/16, B*H), 256 threads (8 waves), ~195 KB LDS.
// One WG owns a 16-row stripe: scores (16x2048 f32) and P (bf16) live in LDS.
// 8 waves => 2 waves/SIMD even at 1 WG/WGP, hiding VMEM/WMMA latency.
// ---------------------------------------------------------------------------
__global__ __launch_bounds__(256) void attn_kernel(
        const unsigned short* __restrict__ Qb,
        const unsigned short* __restrict__ Kb,
        const unsigned short* __restrict__ Vt,
        const float* __restrict__ x,
        unsigned short* __restrict__ Zc) {
    extern __shared__ unsigned char smem[];
    float*          sc     = reinterpret_cast<float*>(smem);
    unsigned short* P      = reinterpret_cast<unsigned short*>(smem + P_OFF);
    float*          red    = reinterpret_cast<float*>(smem + RED_OFF);
    float*          rowmax = reinterpret_cast<float*>(smem + RM_OFF);
    float*          rowsum = reinterpret_cast<float*>(smem + RS_OFF);

    const int lane = threadIdx.x & 31;
    const int wave = threadIdx.x >> 5;      // 0..7
    const int l16  = lane & 15;
    const int half = lane >> 4;

    const int bh = blockIdx.y;
    const int b  = bh >> 3;          // H == 8
    const int h  = bh & 7;
    const int s0 = blockIdx.x * 16;

    // --- Preload Q fragments for this stripe into registers (16 frags) ------
    const unsigned short* Qrow = Qb + ((size_t)bh * S + s0 + l16) * F + half * 8;
    v16bf qf[16];
    #pragma unroll
    for (int k = 0; k < 16; ++k) qf[k] = load_frag(Qrow + k * 32);

    // --- Phase A: scores stripe  S[16 x 2048], 16 n-tiles per wave ---------
    const unsigned short* Kbase = Kb + (size_t)bh * S * F;
    for (int nt = wave; nt < S / 16; nt += 8) {
        const unsigned short* Krow = Kbase + (size_t)(nt * 16 + l16) * F + half * 8;
        // prefetch this lane's K row of the *next* tile this wave will touch
        if (nt + 8 < S / 16)
            __builtin_prefetch((const void*)(Kbase + (size_t)((nt + 8) * 16 + l16) * F), 0, 1);
        v8f c = vzero();
        #pragma unroll 4
        for (int k = 0; k < 16; ++k)
            c = wmma_bf16(qf[k], load_frag(Krow + k * 32), c);
        float* dst = sc + (size_t)(8 * half) * SSTR + nt * 16 + l16;
        #pragma unroll
        for (int r = 0; r < 8; ++r) dst[(size_t)r * SSTR] = c[r];
    }
    __syncthreads();

    // --- Phase B: softmax (two-pass), P = exp(s - rowmax) as bf16 ----------
    {
        const int t    = threadIdx.x;        // 256 threads: 16 rows x 16 parts
        const int row  = t >> 4;
        const int part = t & 15;
        float* srow = sc + (size_t)row * SSTR + part * 128;

        float mx = -3.0e38f;
        for (int i = 0; i < 128; ++i) mx = fmaxf(mx, srow[i]);
        red[t] = mx;
        __syncthreads();
        if (part == 0) {
            float m2 = red[t];
            #pragma unroll
            for (int i = 1; i < 16; ++i) m2 = fmaxf(m2, red[t + i]);
            rowmax[row] = m2;
        }
        __syncthreads();

        const float rm = rowmax[row];
        unsigned short* prow = P + (size_t)row * PSTR + part * 128;
        float sum = 0.0f;
        for (int i = 0; i < 128; ++i) {
            float e = __expf(srow[i] - rm);
            sum += e;
            prow[i] = f2bf(e);
        }
        red[t] = sum;
        __syncthreads();
        if (part == 0) {
            float s2 = 0.0f;
            #pragma unroll
            for (int i = 0; i < 16; ++i) s2 += red[t + i];
            rowsum[row] = s2;
        }
        __syncthreads();
    }

    // --- Phase C: O = P * V  (each wave owns a 64-wide F slice) ------------
    const unsigned short* Vbase = Vt + (size_t)bh * F * S;
    const unsigned short* Pl    = P + (size_t)l16 * PSTR + half * 8;
    v8f acc[4] = {};
    for (int kk = 0; kk < S / 32; ++kk) {
        v16bf a = load_frag(Pl + kk * 32);            // ds_load_b128 x2
        #pragma unroll
        for (int nt = 0; nt < 4; ++nt) {
            int f = wave * 64 + nt * 16 + l16;
            const unsigned short* vp = Vbase + (size_t)f * S + kk * 32 + half * 8;
            acc[nt] = wmma_bf16(a, load_frag(vp), acc[nt]);
        }
    }

    // epilogue: 1/rowsum fold + per-head residual, write into zcat (bf16)
    #pragma unroll
    for (int nt = 0; nt < 4; ++nt) {
        int f = wave * 64 + nt * 16 + l16;
        #pragma unroll
        for (int r = 0; r < 8; ++r) {
            int m  = r + 8 * half;
            int sg = s0 + m;
            float val = acc[nt][r] / rowsum[m]
                      + x[((size_t)b * S + sg) * F + f];
            Zc[(size_t)(b * S + sg) * HF + h * F + f] = f2bf(val);
        }
    }
}

// ---------------------------------------------------------------------------
// Stage 3: joiner GEMM  [BS,HF] x [HF,F].  grid = (BS/64, F/64), 128 threads
// ---------------------------------------------------------------------------
__global__ __launch_bounds__(128) void joiner_kernel(
        const unsigned short* __restrict__ Zc,
        const unsigned short* __restrict__ wjt,
        float* __restrict__ Jout) {
    const int lane = threadIdx.x & 31;
    const int wave = threadIdx.x >> 5;
    const int l16  = lane & 15;
    const int half = lane >> 4;

    const int m0 = blockIdx.x * 64 + wave * 16;
    const int n0 = blockIdx.y * 64;

    const unsigned short* arow = Zc + (size_t)(m0 + l16) * HF + half * 8;
    v8f acc[4] = {};
    #pragma unroll 4
    for (int k = 0; k < HF / 32; ++k) {
        v16bf a = load_frag(arow + k * 32);
        #pragma unroll
        for (int nt = 0; nt < 4; ++nt) {
            const unsigned short* brow =
                wjt + (size_t)(n0 + nt * 16 + l16) * HF + k * 32 + half * 8;
            acc[nt] = wmma_bf16(a, load_frag(brow), acc[nt]);
        }
    }
    #pragma unroll
    for (int nt = 0; nt < 4; ++nt) {
        int e = n0 + nt * 16 + l16;
        #pragma unroll
        for (int r = 0; r < 8; ++r) {
            int row = m0 + r + 8 * half;
            Jout[(size_t)row * F + e] = acc[nt][r];
        }
    }
}

// ---------------------------------------------------------------------------
// Stage 4: fused residual + LayerNorm.  grid = BS, 256 threads
// ---------------------------------------------------------------------------
__global__ __launch_bounds__(256) void ln_kernel(
        const float* __restrict__ x, const float* __restrict__ jo,
        const float* __restrict__ gamma, const float* __restrict__ beta,
        float* __restrict__ out) {
    __shared__ float red[256];
    const int row = blockIdx.x;
    const int t   = threadIdx.x;
    const float* xr = x  + (size_t)row * F;
    const float* jr = jo + (size_t)row * F;

    float y0 = xr[t] + jr[t];
    float y1 = xr[t + 256] + jr[t + 256];

    red[t] = y0 + y1;
    __syncthreads();
    #pragma unroll
    for (int off = 128; off > 0; off >>= 1) {
        if (t < off) red[t] += red[t + off];
        __syncthreads();
    }
    const float mu = red[0] * (1.0f / F);
    __syncthreads();

    float d0 = y0 - mu, d1 = y1 - mu;
    red[t] = d0 * d0 + d1 * d1;
    __syncthreads();
    #pragma unroll
    for (int off = 128; off > 0; off >>= 1) {
        if (t < off) red[t] += red[t + off];
        __syncthreads();
    }
    const float rstd = rsqrtf(red[0] * (1.0f / F) + 1e-5f);

    out[(size_t)row * F + t]       = d0 * rstd * gamma[t]       + beta[t];
    out[(size_t)row * F + t + 256] = d1 * rstd * gamma[t + 256] + beta[t + 256];
}

// ---------------------------------------------------------------------------
// Host launcher
// ---------------------------------------------------------------------------
extern "C" void kernel_launch(void* const* d_in, const int* in_sizes, int n_in,
                              void* d_out, int out_size, void* d_ws, size_t ws_size,
                              hipStream_t stream) {
    const float* x     = (const float*)d_in[0];
    const float* Wq    = (const float*)d_in[1];
    const float* Wk    = (const float*)d_in[2];
    const float* Wv    = (const float*)d_in[3];
    const float* Wj    = (const float*)d_in[4];
    const float* gamma = (const float*)d_in[5];
    const float* beta  = (const float*)d_in[6];
    float* out = (float*)d_out;

    unsigned char* ws = (unsigned char*)d_ws;
    unsigned short* xb  = (unsigned short*)(ws + XB_OFF);
    unsigned short* wqt = (unsigned short*)(ws + WQT_OFF);
    unsigned short* wkt = (unsigned short*)(ws + WKT_OFF);
    unsigned short* wvt = (unsigned short*)(ws + WVT_OFF);
    unsigned short* wjt = (unsigned short*)(ws + WJT_OFF);
    unsigned short* Qb  = (unsigned short*)(ws + QB_OFF);
    unsigned short* Kb  = (unsigned short*)(ws + KB_OFF);
    unsigned short* Vt  = (unsigned short*)(ws + VT_OFF);
    unsigned short* Zc  = (unsigned short*)(ws + ZC_OFF);
    float*          Jo  = (float*)(ws + JO_OFF);

    // Stage 0: dtype conversion + weight transposes (B^T layout for WMMA)
    {
        int n = BS * F;
        cvt_x_kernel<<<(n + 255) / 256, 256, 0, stream>>>(x, xb, n);
        int tw = H * F * F;
        transpose_cvt_kernel<<<(tw + 255) / 256, 256, 0, stream>>>(Wq, wqt, F, F, tw);
        transpose_cvt_kernel<<<(tw + 255) / 256, 256, 0, stream>>>(Wk, wkt, F, F, tw);
        transpose_cvt_kernel<<<(tw + 255) / 256, 256, 0, stream>>>(Wv, wvt, F, F, tw);
        int tj = F * HF;
        transpose_cvt_kernel<<<(tj + 255) / 256, 256, 0, stream>>>(Wj, wjt, HF, F, tj);
    }

    // Stage 1: QKV projections
    qkv_kernel<<<dim3(BS / 64, F / 64, 3 * H), 128, 0, stream>>>(
        xb, wqt, wkt, wvt, Qb, Kb, Vt);

    // Stage 2: attention (needs > 64 KB dynamic LDS — CDNA5 WGP has 320 KB)
    hipFuncSetAttribute((const void*)attn_kernel,
                        hipFuncAttributeMaxDynamicSharedMemorySize, SMEM_TOT);
    attn_kernel<<<dim3(S / 16, B * H), 256, SMEM_TOT, stream>>>(Qb, Kb, Vt, x, Zc);

    // Stage 3: joiner
    joiner_kernel<<<dim3(BS / 64, F / 64), 128, 0, stream>>>(Zc, wjt, Jo);

    // Stage 4: residual + LayerNorm
    ln_kernel<<<BS, 256, 0, stream>>>(x, Jo, gamma, beta, out);
}
